// DLSAGE_54984171323620
// MI455X (gfx1250) — matrix operand order repaired and checked by reading, not compiled, not run on hardware
//
#include <hip/hip_runtime.h>

typedef unsigned short u16;
typedef __attribute__((ext_vector_type(16))) __bf16 v16bf;
typedef __attribute__((ext_vector_type(8)))  float  v8f;

union FragU {
    uint4 q[2];
    v16bf v;
};

__device__ __forceinline__ u16 f2bf(float f) {
    unsigned u = __float_as_uint(f);
    unsigned r = u + 0x7fffu + ((u >> 16) & 1u);   // round-to-nearest-even
    return (u16)(r >> 16);
}
__device__ __forceinline__ float bf2f(u16 h) {
    return __uint_as_float(((unsigned)h) << 16);
}

// ---------------------------------------------------------------------------
// Pack an f32 [K x N] row-major weight matrix into per-lane WMMA B-fragment
// order: dst[(((kt*ntiles + tn)*32 + lane)*16 + j]
//   lane<16 : col = tn*16+lane,     k = kt*32 +     {0-7,16-23}[j]   (j = 0..15)
//   lane>=16: col = tn*16+lane-16,  k = kt*32 + {8-15,24-31}[j]
// Out-of-range (k>=K || col>=N) -> 0 (zero padding).
// ---------------------------------------------------------------------------
__global__ void sage_pack_b(const float* __restrict__ src, int K, int N,
                            int ktiles, int ntiles, u16* __restrict__ dst) {
    int idx = blockIdx.x * blockDim.x + threadIdx.x;
    int total = ktiles * ntiles * 32 * 16;
    if (idx >= total) return;
    int j    = idx & 15;
    int lane = (idx >> 4) & 31;
    int t    = idx >> 9;
    int tn   = t % ntiles;
    int kt   = t / ntiles;
    int col  = tn * 16 + (lane & 15);
    int k    = kt * 32 + ((lane < 16) ? 0 : 8) + (j & 7) + ((j & 8) ? 16 : 0);
    float v  = (k < K && col < N) ? src[(size_t)k * N + col] : 0.0f;
    dst[idx] = f2bf(v);
}

__global__ void sage_cvt_bf16(const float* __restrict__ src, u16* __restrict__ dst, size_t n) {
    size_t i = (size_t)blockIdx.x * blockDim.x + threadIdx.x;
    size_t s = (size_t)gridDim.x * blockDim.x;
    for (; i < n; i += s) dst[i] = f2bf(src[i]);
}

__global__ void sage_zero_f32(float* __restrict__ p, size_t n) {
    size_t i = (size_t)blockIdx.x * blockDim.x + threadIdx.x;
    size_t s = (size_t)gridDim.x * blockDim.x;
    for (; i < n; i += s) p[i] = 0.0f;
}

// zero cols 240..255 of the lda=256 encoder activation buffer
__global__ void sage_zero_pad240(u16* __restrict__ h, int M) {
    int r = blockIdx.x * blockDim.x + threadIdx.x;
    if (r >= M) return;
    uint4 z = make_uint4(0u, 0u, 0u, 0u);
    uint4* p = (uint4*)(h + (size_t)r * 256 + 240);
    p[0] = z;
    p[1] = z;
}

__global__ void sage_degree(const int* __restrict__ dst, float* __restrict__ deg, int E) {
    int e = blockIdx.x * blockDim.x + threadIdx.x;
    if (e >= E) return;
    atomicAdd(&deg[dst[e]], 1.0f);
}

// gather h[src] (24 bf16, padded row stride 32) and scatter-add f32 into agg[dst]
__global__ void sage_scatter(const u16* __restrict__ h, const int* __restrict__ src,
                             const int* __restrict__ dst, float* __restrict__ agg, int E) {
    int e = blockIdx.x * blockDim.x + threadIdx.x;
    if (e >= E) return;
    int s = src[e];
    int d = dst[e];
    const uint4* hr = (const uint4*)(h + (size_t)s * 32);
    uint4 q0 = hr[0], q1 = hr[1], q2 = hr[2];
    unsigned w[12] = {q0.x, q0.y, q0.z, q0.w, q1.x, q1.y, q1.z, q1.w, q2.x, q2.y, q2.z, q2.w};
    float* ar = agg + (size_t)d * 24;
#pragma unroll
    for (int i = 0; i < 12; ++i) {
        atomicAdd(ar + 2 * i,     __uint_as_float((w[i] & 0xffffu) << 16));
        atomicAdd(ar + 2 * i + 1, __uint_as_float(w[i] & 0xffff0000u));
    }
}

// mean = agg / max(deg,1)  -> bf16 [M x 32], cols 24..31 zeroed
__global__ void sage_mean(const float* __restrict__ agg, const float* __restrict__ deg,
                          u16* __restrict__ meanb, int M) {
    int idx = blockIdx.x * blockDim.x + threadIdx.x;
    if (idx >= M * 32) return;
    int n = idx >> 5;
    int c = idx & 31;
    u16 v = 0;
    if (c < 24) {
        float m = agg[(size_t)n * 24 + c] / fmaxf(deg[n], 1.0f);
        v = f2bf(m);
    }
    meanb[idx] = v;
}

// ---------------------------------------------------------------------------
// Accumulate one (A,Bpacked) pair over ktiles K-tiles into the wave's C frag.
// A: bf16, row-major, padded leading dim lda (>= ktiles*32), 16B-aligned rows.
// 16-bit A frag layout (ISA 7.12.2): lane<16 row=lane holds K {0-7,16-23};
// lane>=16 row=lane-16 holds K {8-15,24-31}  -> two b128 loads per lane.
// ---------------------------------------------------------------------------
__device__ __forceinline__ v8f sage_acc(v8f c, const u16* __restrict__ A, int lda,
                                        const u16* __restrict__ Bp, int ktiles,
                                        int ntiles, int tn, int lane) {
    const int half = (lane < 16) ? 0 : 8;
    const u16* arow = A + (size_t)(lane & 15) * lda;
    for (int kt = 0; kt < ktiles; ++kt) {
        int kb = kt * 32 + half;
        FragU a, b;
        a.q[0] = *(const uint4*)(arow + kb);
        a.q[1] = *(const uint4*)(arow + kb + 16);
        const u16* bl = Bp + (((size_t)kt * ntiles + tn) * 32 + lane) * 16;
        b.q[0] = *(const uint4*)(bl);
        b.q[1] = *(const uint4*)(bl + 8);
        c = __builtin_amdgcn_wmma_f32_16x16x32_bf16(false, a.v, false, b.v,
                                                    (short)0, c, false, false);
    }
    return c;
}

// ---------------------------------------------------------------------------
// Generic tall-skinny GEMM: one wave per 16x16 output tile, 8 waves/block.
// C = relu?( A1@B1 [+ A2@B2] + bias ), written as bf16 (outb) and/or f32 (outf).
// ---------------------------------------------------------------------------
__global__ void sage_gemm_wmma(const u16* __restrict__ A1, int lda1,
                               const u16* __restrict__ B1, int kt1,
                               const u16* __restrict__ A2, int lda2,
                               const u16* __restrict__ B2, int kt2,
                               int mtiles, int ntiles,
                               const float* __restrict__ bias, int nvalid, int relu,
                               u16* __restrict__ outb, int ldob,
                               float* __restrict__ outf, int ldof, int nfvalid) {
    int wave = (blockIdx.x * blockDim.x + threadIdx.x) >> 5;
    int lane = threadIdx.x & 31;
    if (wave >= mtiles * ntiles) return;   // wave-uniform; EXEC stays all-1s for WMMA
    int tn = wave % ntiles;
    int tm = wave / ntiles;

    v8f c = {};
    c = sage_acc(c, A1 + (size_t)tm * 16 * lda1, lda1, B1, kt1, ntiles, tn, lane);
    if (A2)
        c = sage_acc(c, A2 + (size_t)tm * 16 * lda2, lda2, B2, kt2, ntiles, tn, lane);

    // C/D layout: VGPR r -> (row tm*16 + r + (lane<16?0:8), col tn*16 + (lane&15))
    int col  = tn * 16 + (lane & 15);
    float bv = (bias && col < nvalid) ? bias[col] : 0.0f;
    int row0 = tm * 16 + ((lane < 16) ? 0 : 8);
#pragma unroll
    for (int r = 0; r < 8; ++r) {
        float v = c[r] + bv;
        if (relu) v = fmaxf(v, 0.0f);
        if (outb) outb[(size_t)(row0 + r) * ldob + col] = f2bf(v);
        if (outf && col < nfvalid) outf[(size_t)(row0 + r) * ldof + col] = v;
    }
}

// ---------------------------------------------------------------------------
extern "C" void kernel_launch(void* const* d_in, const int* in_sizes, int n_in,
                              void* d_out, int out_size, void* d_ws, size_t ws_size,
                              hipStream_t stream) {
    const float* x      = (const float*)d_in[0];
    const int*   esrc   = (const int*)d_in[1];
    const int    E      = in_sizes[1] / 2;
    const int*   edst   = esrc + E;
    const float* enc_w1 = (const float*)d_in[2];
    const float* enc_b1 = (const float*)d_in[3];
    const float* enc_w2 = (const float*)d_in[4];
    const float* enc_b2 = (const float*)d_in[5];
    const float* s_wl[3] = {(const float*)d_in[6],  (const float*)d_in[9],  (const float*)d_in[12]};
    const float* s_bl[3] = {(const float*)d_in[7],  (const float*)d_in[10], (const float*)d_in[13]};
    const float* s_wr[3] = {(const float*)d_in[8],  (const float*)d_in[11], (const float*)d_in[14]};
    const float* dec_w1 = (const float*)d_in[15];
    const float* dec_b1 = (const float*)d_in[16];
    const float* dec_w2 = (const float*)d_in[17];
    const float* dec_b2 = (const float*)d_in[18];
    float* out = (float*)d_out;

    const int M      = in_sizes[0] / 32;   // 100000, multiple of 16
    const int mtiles = M / 16;

    // ---- workspace carve-up (256B aligned chunks) ----
    size_t off = 0;
    auto carve = [&](size_t bytes) -> void* {
        void* p = (char*)d_ws + off;
        off += (bytes + 255) & ~(size_t)255;
        return p;
    };
    u16*   xbf   = (u16*)  carve((size_t)M * 32 * 2);     // x in bf16
    u16*   h240  = (u16*)  carve((size_t)M * 256 * 2);    // encoder hidden (lda 256); reused for h120
    u16*   hA    = (u16*)  carve((size_t)M * 32 * 2);     // H=24 ping
    u16*   hB    = (u16*)  carve((size_t)M * 32 * 2);     // H=24 pong
    u16*   meanb = (u16*)  carve((size_t)M * 32 * 2);     // mean-aggregated, bf16
    float* agg   = (float*)carve((size_t)M * 24 * 4);
    float* deg   = (float*)carve((size_t)M * 4);
    u16*   p_e1  = (u16*)  carve((size_t)1 * 15 * 512 * 2);  // enc_w1  K32  N240
    u16*   p_e2  = (u16*)  carve((size_t)8 * 2  * 512 * 2);  // enc_w2  K256 N32
    u16*   p_wl[3], *p_wr[3];
    for (int i = 0; i < 3; ++i) {
        p_wl[i] = (u16*)carve((size_t)1 * 2 * 512 * 2);      // K32 N32
        p_wr[i] = (u16*)carve((size_t)1 * 2 * 512 * 2);
    }
    u16*   p_d1  = (u16*)  carve((size_t)1 * 8 * 512 * 2);   // dec_w1 K32  N128
    u16*   p_d2  = (u16*)  carve((size_t)4 * 1 * 512 * 2);   // dec_w2 K128 N16
    u16*   h120  = h240;                                      // alias (h240 dead by then)
    (void)ws_size; (void)n_in; (void)out_size;

    const int B = 256;
    auto blocks1d = [&](size_t n) { return (int)((n + B - 1) / B); };

    // ---- pack weights to WMMA fragment order (cheap, once per call) ----
    auto pack = [&](const float* src, int K, int N, int kt, int nt, u16* dst) {
        int total = kt * nt * 512;
        sage_pack_b<<<blocks1d(total), B, 0, stream>>>(src, K, N, kt, nt, dst);
    };
    pack(enc_w1, 32, 240, 1, 15, p_e1);
    pack(enc_w2, 240, 24, 8, 2,  p_e2);
    for (int i = 0; i < 3; ++i) {
        pack(s_wl[i], 24, 24, 1, 2, p_wl[i]);
        pack(s_wr[i], 24, 24, 1, 2, p_wr[i]);
    }
    pack(dec_w1, 24, 120, 1, 8, p_d1);
    pack(dec_w2, 120, 12, 4, 1, p_d2);

    // ---- input conversion + degrees ----
    sage_cvt_bf16<<<4096, B, 0, stream>>>(x, xbf, (size_t)M * 32);
    sage_zero_f32<<<blocks1d(M), B, 0, stream>>>(deg, (size_t)M);
    sage_degree<<<blocks1d(E), B, 0, stream>>>(edst, deg, E);

    auto gemm = [&](const u16* A1, int lda1, const u16* B1, int kt1,
                    const u16* A2, int lda2, const u16* B2, int kt2,
                    int ntiles, const float* bias, int nvalid, int relu,
                    u16* outb, int ldob, float* outf, int ldof, int nfv) {
        int waves  = mtiles * ntiles;
        int blocks = (waves + 7) / 8;     // 8 waves per 256-thread block
        sage_gemm_wmma<<<blocks, B, 0, stream>>>(A1, lda1, B1, kt1, A2, lda2, B2, kt2,
                                                 mtiles, ntiles, bias, nvalid, relu,
                                                 outb, ldob, outf, ldof, nfv);
    };

    // ---- encoder: [M,32]@[32,240] relu -> [M,240(256)]@[240,24] relu -> hA ----
    sage_zero_pad240<<<blocks1d(M), B, 0, stream>>>(h240, M);
    gemm(xbf, 32, p_e1, 1, nullptr, 0, nullptr, 0, 15, enc_b1, 240, 1,
         h240, 256, nullptr, 0, 0);
    gemm(h240, 256, p_e2, 8, nullptr, 0, nullptr, 0, 2, enc_b2, 24, 1,
         hA, 32, nullptr, 0, 0);

    // ---- 3x SAGEConv: mean-aggregate then fused (mean@wl + h@wr + bl) ----
    u16* hin  = hA;
    u16* hout = hB;
    for (int L = 0; L < 3; ++L) {
        sage_zero_f32<<<4096, B, 0, stream>>>(agg, (size_t)M * 24);
        sage_scatter<<<blocks1d(E), B, 0, stream>>>(hin, esrc, edst, agg, E);
        sage_mean<<<blocks1d((size_t)M * 32), B, 0, stream>>>(agg, deg, meanb, M);
        gemm(meanb, 32, p_wl[L], 1, hin, 32, p_wr[L], 1, 2, s_bl[L], 24,
             (L < 2) ? 1 : 0, hout, 32, nullptr, 0, 0);
        u16* t = hin; hin = hout; hout = t;
    }

    // ---- decoder: [M,24]@[24,120] relu -> [M,120(128)]@[120,12] -> d_out ----
    gemm(hin, 32, p_d1, 1, nullptr, 0, nullptr, 0, 8, dec_b1, 120, 1,
         h120, 128, nullptr, 0, 0);
    gemm(h120, 128, p_d2, 4, nullptr, 0, nullptr, 0, 1, dec_b2, 12, 0,
         nullptr, 0, out, 12, 12);
}
